// HierarchicalPrunedAttentionNetwork_13134009991848
// MI455X (gfx1250) — compile-verified, matrix operand order checked by reading, not run on hardware
//
#include <hip/hip_runtime.h>
#include <hip/hip_bf16.h>

// ---------------------------------------------------------------------------
// CDNA5 (gfx1250) WMMA helpers: bf16 16x16x32, f32 accumulate, wave32.
// All GEMMs are  C[M,N] = X[M,K] * W[N,K]^T  with X and W row-major, so the
// A fragment (rows of X) and B fragment (rows of W == columns of B) use the
// identical per-lane layout:
//   lane (l<16)  : row l,   K elements {k0+0..7,  k0+16..23}
//   lane (l>=16) : row l-16,K elements {k0+8..15, k0+24..31}
// i.e. two contiguous 16-byte chunks per lane per 32-wide K step.
// C/D layout (f32, 8 VGPRs): lane l holds column n=l%16, rows m=8*(l/16)+v.
// ---------------------------------------------------------------------------

typedef __attribute__((ext_vector_type(16))) __bf16 v16bf;
typedef __attribute__((ext_vector_type(8)))  float  v8f;

union FragU { v16bf v; uint4 q[2]; };

__device__ __forceinline__ v16bf load_frag(const __hip_bfloat16* base, int ld,
                                           int row, int k0, int lane) {
    int half = lane >> 4;
    const __hip_bfloat16* p = base + (size_t)row * ld + k0 + 8 * half;
    FragU f;
    f.q[0] = *reinterpret_cast<const uint4*>(p);
    f.q[1] = *reinterpret_cast<const uint4*>(p + 16);
    return f.v;
}

__device__ __forceinline__ v8f wmma_bf16(v16bf a, v16bf b, v8f c) {
    return __builtin_amdgcn_wmma_f32_16x16x32_bf16(
        /*neg_a=*/false, a, /*neg_b=*/false, b,
        /*c_mod=*/(short)0, c, /*reuse_a=*/false, /*reuse_b=*/false);
}

__device__ __forceinline__ float sigmoidf_(float x) {
    return 1.0f / (1.0f + __expf(-x));
}

// ---------------------------------------------------------------------------
// Utility kernels
// ---------------------------------------------------------------------------

__global__ void zero_u32(unsigned* __restrict__ p, int n) {
    int i = blockIdx.x * blockDim.x + threadIdx.x;
    if (i < n) p[i] = 0u;
}

// f32 [N,K] row-major -> bf16 [N,Kp] row-major (zero padded K -> Kp)
__global__ void cvt_pad_bf16(const float* __restrict__ src,
                             __hip_bfloat16* __restrict__ dst,
                             int N, int K, int Kp) {
    int i = blockIdx.x * blockDim.x + threadIdx.x;
    int total = N * Kp;
    if (i >= total) return;
    int r = i / Kp, c = i - r * Kp;
    dst[i] = __float2bfloat16(c < K ? src[(size_t)r * K + c] : 0.0f);
}

// Embedding gather: tokens [65536] -> words_bf [65536, 320] (E=300 padded)
__global__ void gather_embed(const int* __restrict__ tokens,
                             const float* __restrict__ emb,
                             __hip_bfloat16* __restrict__ out) {
    int row = blockIdx.x;
    int c   = threadIdx.x;              // 0..319
    int tok = tokens[row];
    float v = (c < 300) ? emb[(size_t)tok * 300 + c] : 0.0f;
    out[(size_t)row * 320 + c] = __float2bfloat16(v);
}

// ---------------------------------------------------------------------------
// Generic WMMA GEMM: out[M,ldo] = X[M,ldx] * W[N,ldw]^T (+ bias[n])
// One wave per 16x16 output tile. outF (f32) and/or outB (bf16) may be null.
// ---------------------------------------------------------------------------
__global__ __launch_bounds__(32)
void gemm_xwt_bf16(const __hip_bfloat16* __restrict__ X, int ldx,
                   const __hip_bfloat16* __restrict__ W, int ldw,
                   const float* __restrict__ bias,
                   float* __restrict__ outF,
                   __hip_bfloat16* __restrict__ outB,
                   int ldo, int K) {
    int lane = threadIdx.x;
    int n0 = blockIdx.x * 16;
    int m0 = blockIdx.y * 16;
    v8f acc = {};
    int ra = m0 + (lane & 15);
    int rb = n0 + (lane & 15);
    for (int k0 = 0; k0 < K; k0 += 32) {
        v16bf a = load_frag(X, ldx, ra, k0, lane);
        v16bf b = load_frag(W, ldw, rb, k0, lane);
        acc = wmma_bf16(a, b, acc);
    }
    int c = n0 + (lane & 15);
    float bv = bias ? bias[c] : 0.0f;
    int mbase = m0 + 8 * (lane >> 4);
    for (int v = 0; v < 8; ++v) {
        float val = acc[v] + bv;
        size_t idx = (size_t)(mbase + v) * ldo + c;
        if (outF) outF[idx] = val;
        if (outB) outB[idx] = __float2bfloat16(val);
    }
}

// ---------------------------------------------------------------------------
// GRU step: one kernel launch per timestep (grid-wide dependency).
//   gh = h_prev @ Wh^T (three 16x16 WMMA accumulators: r, z, n gates)
//   r = sig(gi_r + gh_r + bh_r); z = sig(gi_z + gh_z + bh_z)
//   n = tanh(gi_n + r*(gh_n + bh_n)); h = (1-z)*n + z*h_prev
// grid = (16 col-tiles, NB/16 seq-tiles), block = 32 (one wave).
// ---------------------------------------------------------------------------
__global__ __launch_bounds__(32)
void gru_step(const __hip_bfloat16* __restrict__ hin_bf,   // [NB,256]
              const float*          __restrict__ hin_f,    // [NB,256]
              __hip_bfloat16*       __restrict__ hout_bf,  // [NB,256]
              float*                __restrict__ hout_f,   // [NB,256]
              const __hip_bfloat16* __restrict__ Wh,       // [768,256]
              const float*          __restrict__ bh,       // [768]
              const __hip_bfloat16* __restrict__ gi,       // [NB*T,768]
              float*                __restrict__ hid,      // [NB*T,512]
              __hip_bfloat16*       __restrict__ hid_bf,   // [NB*T,512]
              int T, int t, int col_off) {
    int lane = threadIdx.x;
    int c0 = blockIdx.x * 16;
    int m0 = blockIdx.y * 16;
    v8f accR = {}, accZ = {}, accN = {};
    int ra = m0 + (lane & 15);
    int rb = c0 + (lane & 15);
    for (int k0 = 0; k0 < 256; k0 += 32) {
        v16bf a  = load_frag(hin_bf, 256, ra, k0, lane);
        v16bf bR = load_frag(Wh, 256, rb,        k0, lane);
        v16bf bZ = load_frag(Wh, 256, rb + 256,  k0, lane);
        v16bf bN = load_frag(Wh, 256, rb + 512,  k0, lane);
        accR = wmma_bf16(a, bR, accR);
        accZ = wmma_bf16(a, bZ, accZ);
        accN = wmma_bf16(a, bN, accN);
    }
    int c = c0 + (lane & 15);
    float bhr = bh[c], bhz = bh[256 + c], bhn = bh[512 + c];
    int mbase = m0 + 8 * (lane >> 4);
    for (int v = 0; v < 8; ++v) {
        int seq = mbase + v;
        size_t row = (size_t)seq * T + t;
        const __hip_bfloat16* g = gi + row * 768;
        float gr = __bfloat162float(g[c]);
        float gz = __bfloat162float(g[256 + c]);
        float gn = __bfloat162float(g[512 + c]);
        float hp = hin_f[(size_t)seq * 256 + c];
        float r = sigmoidf_(gr + accR[v] + bhr);
        float z = sigmoidf_(gz + accZ[v] + bhz);
        float n = tanhf(gn + r * (accN[v] + bhn));
        float h = (1.0f - z) * n + z * hp;
        hout_f[(size_t)seq * 256 + c] = h;
        hout_bf[(size_t)seq * 256 + c] = __float2bfloat16(h);
        size_t oi = row * 512 + col_off + c;
        hid[oi] = h;
        hid_bf[oi] = __float2bfloat16(h);
    }
}

// ---------------------------------------------------------------------------
// Fused attention scoring: score[m] += sum_n tanh((H@Wrep^T)[m,n] + b[n])*ctx[n]
// Avoids materializing the [M,512] "rep" tensor. 16-lane tree reduce across
// each C-column group, then atomicAdd per row.
// ---------------------------------------------------------------------------
__global__ __launch_bounds__(32)
void rep_score(const __hip_bfloat16* __restrict__ Hbf,   // [M,512]
               const __hip_bfloat16* __restrict__ Wr,    // [512,512]
               const float* __restrict__ br,             // [512]
               const float* __restrict__ ctx,            // [512]
               float* __restrict__ scores) {             // [M]
    int lane = threadIdx.x;
    int n0 = blockIdx.x * 16;
    int m0 = blockIdx.y * 16;
    v8f acc = {};
    int ra = m0 + (lane & 15);
    int rb = n0 + (lane & 15);
    for (int k0 = 0; k0 < 512; k0 += 32) {
        v16bf a = load_frag(Hbf, 512, ra, k0, lane);
        v16bf b = load_frag(Wr, 512, rb, k0, lane);
        acc = wmma_bf16(a, b, acc);
    }
    int c = n0 + (lane & 15);
    float bv = br[c], cv = ctx[c];
    int mbase = m0 + 8 * (lane >> 4);
    for (int v = 0; v < 8; ++v) {
        float s = tanhf(acc[v] + bv) * cv;
        // reduce across the 16 lanes holding this row's 16 columns
        for (int m = 1; m < 16; m <<= 1) s += __shfl_xor(s, m, 32);
        if ((lane & 15) == 0) atomicAdd(scores + mbase + v, s);
    }
}

// ---------------------------------------------------------------------------
// Softmax -> threshold prune (keep w < 0.15) -> renorm -> weighted pool.
// One block per sequence (T <= 64), 256 threads cover 512 hidden columns.
// ---------------------------------------------------------------------------
__global__ void attend_finalize(const float* __restrict__ hid,    // [NS*T,512]
                                const float* __restrict__ scores, // [NS*T]
                                float* __restrict__ outF,         // [NS,512]
                                __hip_bfloat16* __restrict__ outB,
                                int T) {
    __shared__ float sw[64];
    __shared__ float sb[64];
    int seq = blockIdx.x, tid = threadIdx.x;
    if (tid < T) sw[tid] = scores[(size_t)seq * T + tid];
    __syncthreads();
    if (tid == 0) {
        float mx = -3.4e38f;
        for (int i = 0; i < T; ++i) mx = fmaxf(mx, sw[i]);
        float sum = 0.0f;
        for (int i = 0; i < T; ++i) { float e = __expf(sw[i] - mx); sb[i] = e; sum += e; }
        float ps = 0.0f;
        for (int i = 0; i < T; ++i) {
            float w = sb[i] / sum;
            w = (w < 0.15f) ? w : 0.0f;     // keep weights strictly below threshold
            sb[i] = w; ps += w;
        }
        float inv = (ps > 0.0f) ? 1.0f / ps : 0.0f;   // NaN cleanup -> 0
        for (int i = 0; i < T; ++i) sw[i] = sb[i] * inv;
    }
    __syncthreads();
    float a0 = 0.0f, a1 = 0.0f;
    for (int i = 0; i < T; ++i) {
        float p = sw[i];
        const float* h = hid + ((size_t)seq * T + i) * 512;
        a0 += p * h[tid];
        a1 += p * h[tid + 256];
    }
    size_t o = (size_t)seq * 512 + tid;
    outF[o] = a0; outF[o + 256] = a1;
    if (outB) {
        outB[o] = __float2bfloat16(a0);
        outB[o + 256] = __float2bfloat16(a1);
    }
}

// Final classifier: out[32,5] = doc[32,512] @ out_W[5,512]^T + out_b
__global__ void out_linear(const float* __restrict__ doc,
                           const float* __restrict__ W,
                           const float* __restrict__ b,
                           float* __restrict__ out) {
    int i = threadIdx.x;
    if (i >= 160) return;
    int bb = i / 5, c = i - bb * 5;
    const float* d = doc + (size_t)bb * 512;
    const float* w = W + (size_t)c * 512;
    float s = b[c];
    for (int k = 0; k < 512; ++k) s += d[k] * w[k];
    out[i] = s;
}

// ---------------------------------------------------------------------------
// Host orchestration
// ---------------------------------------------------------------------------
extern "C" void kernel_launch(void* const* d_in, const int* in_sizes, int n_in,
                              void* d_out, int out_size, void* d_ws, size_t ws_size,
                              hipStream_t stream) {
    (void)in_sizes; (void)n_in; (void)out_size; (void)ws_size;

    const int*   tokens = (const int*)  d_in[0];   // [32,32,64]
    const float* emb    = (const float*)d_in[1];   // [50000,300]
    const float* wWi    = (const float*)d_in[2];   // [2,768,300]
    const float* wWh    = (const float*)d_in[3];   // [2,768,256]
    const float* wbi    = (const float*)d_in[4];   // [2,768]
    const float* wbh    = (const float*)d_in[5];   // [2,768]
    const float* wrepW  = (const float*)d_in[6];   // [512,512]
    const float* wrepB  = (const float*)d_in[7];   // [512]
    const float* wctx   = (const float*)d_in[8];   // [512]
    const float* sWi    = (const float*)d_in[9];   // [2,768,512]
    const float* sWh    = (const float*)d_in[10];  // [2,768,256]
    const float* sbi    = (const float*)d_in[11];  // [2,768]
    const float* sbh    = (const float*)d_in[12];  // [2,768]
    const float* srepW  = (const float*)d_in[13];  // [512,512]
    const float* srepB  = (const float*)d_in[14];  // [512]
    const float* sctx   = (const float*)d_in[15];  // [512]
    const float* outW   = (const float*)d_in[16];  // [5,512]
    const float* outB   = (const float*)d_in[17];  // [5]
    float* out = (float*)d_out;                    // [32,5]

    const size_t NW = 65536;   // word rows  = 1024 seqs * 64 steps
    const size_t NS = 1024;    // sent rows  = 32 docs * 32 steps
    const int EP = 320;        // 300 padded to multiple of 32

    // workspace layout
    char* ws = (char*)d_ws;
    size_t off = 0;
    auto alloc = [&](size_t bytes) {
        size_t r = off; off += (bytes + 255) & ~(size_t)255; return r;
    };
    size_t o_words  = alloc(NW * EP * 2);                 // words bf16
    size_t o_WiW    = alloc((size_t)2 * 768 * 320 * 2);
    size_t o_WhW    = alloc((size_t)2 * 768 * 256 * 2);
    size_t o_WiS    = alloc((size_t)2 * 768 * 512 * 2);
    size_t o_WhS    = alloc((size_t)2 * 768 * 256 * 2);
    size_t o_wrep   = alloc((size_t)512 * 512 * 2);
    size_t o_srep   = alloc((size_t)512 * 512 * 2);
    size_t o_giW    = alloc((size_t)2 * NW * 768 * 2);    // bf16
    size_t o_whid   = alloc(NW * 512 * 4);
    size_t o_whidbf = alloc(NW * 512 * 2);
    size_t o_hAf    = alloc((size_t)1024 * 256 * 4);
    size_t o_hAb    = alloc((size_t)1024 * 256 * 2);
    size_t o_hBf    = alloc((size_t)1024 * 256 * 4);
    size_t o_hBb    = alloc((size_t)1024 * 256 * 2);
    size_t o_scW    = alloc(NW * 4);
    size_t o_sentf  = alloc(NS * 512 * 4);
    size_t o_sentb  = alloc(NS * 512 * 2);
    size_t o_giS    = alloc((size_t)2 * NS * 768 * 2);
    size_t o_shid   = alloc(NS * 512 * 4);
    size_t o_shidbf = alloc(NS * 512 * 2);
    size_t o_scS    = alloc(NS * 4);
    size_t o_doc    = alloc((size_t)32 * 512 * 4);

    auto bfp = [&](size_t o) { return (__hip_bfloat16*)(ws + o); };
    auto f32p = [&](size_t o) { return (float*)(ws + o); };

    // ---- weight conversions (f32 -> bf16, K padded to multiple of 32) ----
    auto cvt = [&](const float* s, size_t o, int N, int K, int Kp) {
        int total = N * Kp;
        cvt_pad_bf16<<<(total + 255) / 256, 256, 0, stream>>>(s, bfp(o), N, K, Kp);
    };
    cvt(wWi,   o_WiW, 1536, 300, 320);
    cvt(wWh,   o_WhW, 1536, 256, 256);
    cvt(sWi,   o_WiS, 1536, 512, 512);
    cvt(sWh,   o_WhS, 1536, 256, 256);
    cvt(wrepW, o_wrep, 512, 512, 512);
    cvt(srepW, o_srep, 512, 512, 512);

    // ---- embedding gather (padding_idx handled by emb row 0 being zero) ----
    gather_embed<<<(unsigned)NW, 320, 0, stream>>>(tokens, emb, bfp(o_words));

    // ---- word-level input projections: gi = words @ Wi^T + bi (bf16 out) ----
    for (int dir = 0; dir < 2; ++dir) {
        gemm_xwt_bf16<<<dim3(48, 4096), 32, 0, stream>>>(
            bfp(o_words), 320,
            bfp(o_WiW) + (size_t)dir * 768 * 320, 320,
            wbi + (size_t)dir * 768,
            nullptr, bfp(o_giW) + (size_t)dir * NW * 768,
            768, 320);
    }

    // ---- word-level BiGRU recurrence: 64 sequential steps per direction ----
    for (int dir = 0; dir < 2; ++dir) {
        zero_u32<<<(1024 * 256 + 255) / 256, 256, 0, stream>>>((unsigned*)f32p(o_hAf), 1024 * 256);
        zero_u32<<<(1024 * 128 + 255) / 256, 256, 0, stream>>>((unsigned*)(ws + o_hAb), 1024 * 128);
        for (int s = 0; s < 64; ++s) {
            int t = dir ? (63 - s) : s;
            size_t inf = (s & 1) ? o_hBf : o_hAf, inb = (s & 1) ? o_hBb : o_hAb;
            size_t otf = (s & 1) ? o_hAf : o_hBf, otb = (s & 1) ? o_hAb : o_hBb;
            gru_step<<<dim3(16, 64), 32, 0, stream>>>(
                bfp(inb), f32p(inf), bfp(otb), f32p(otf),
                bfp(o_WhW) + (size_t)dir * 768 * 256,
                wbh + (size_t)dir * 768,
                bfp(o_giW) + (size_t)dir * NW * 768,
                f32p(o_whid), bfp(o_whidbf),
                64, t, dir * 256);
        }
    }

    // ---- word attention: fused score, then pooled sentence vectors ----
    zero_u32<<<((int)NW + 255) / 256, 256, 0, stream>>>((unsigned*)f32p(o_scW), (int)NW);
    rep_score<<<dim3(32, 4096), 32, 0, stream>>>(
        bfp(o_whidbf), bfp(o_wrep), wrepB, wctx, f32p(o_scW));
    attend_finalize<<<1024, 256, 0, stream>>>(
        f32p(o_whid), f32p(o_scW), f32p(o_sentf), bfp(o_sentb), 64);

    // ---- sentence-level input projections ----
    for (int dir = 0; dir < 2; ++dir) {
        gemm_xwt_bf16<<<dim3(48, 64), 32, 0, stream>>>(
            bfp(o_sentb), 512,
            bfp(o_WiS) + (size_t)dir * 768 * 512, 512,
            sbi + (size_t)dir * 768,
            nullptr, bfp(o_giS) + (size_t)dir * NS * 768,
            768, 512);
    }

    // ---- sentence-level BiGRU recurrence: 32 steps per direction (NB=32) ----
    for (int dir = 0; dir < 2; ++dir) {
        zero_u32<<<(32 * 256 + 255) / 256, 256, 0, stream>>>((unsigned*)f32p(o_hAf), 32 * 256);
        zero_u32<<<(32 * 128 + 255) / 256, 256, 0, stream>>>((unsigned*)(ws + o_hAb), 32 * 128);
        for (int s = 0; s < 32; ++s) {
            int t = dir ? (31 - s) : s;
            size_t inf = (s & 1) ? o_hBf : o_hAf, inb = (s & 1) ? o_hBb : o_hAb;
            size_t otf = (s & 1) ? o_hAf : o_hBf, otb = (s & 1) ? o_hAb : o_hBb;
            gru_step<<<dim3(16, 2), 32, 0, stream>>>(
                bfp(inb), f32p(inf), bfp(otb), f32p(otf),
                bfp(o_WhS) + (size_t)dir * 768 * 256,
                sbh + (size_t)dir * 768,
                bfp(o_giS) + (size_t)dir * NS * 768,
                f32p(o_shid), bfp(o_shidbf),
                32, t, dir * 256);
        }
    }

    // ---- sentence attention -> doc vectors ----
    zero_u32<<<((int)NS + 255) / 256, 256, 0, stream>>>((unsigned*)f32p(o_scS), (int)NS);
    rep_score<<<dim3(32, 64), 32, 0, stream>>>(
        bfp(o_shidbf), bfp(o_srep), srepB, sctx, f32p(o_scS));
    attend_finalize<<<32, 256, 0, stream>>>(
        f32p(o_shid), f32p(o_scS), f32p(o_doc), nullptr, 32);

    // ---- classifier ----
    out_linear<<<1, 160, 0, stream>>>(f32p(o_doc), outW, outB, out);
}
